// Decoder_76381698392887
// MI455X (gfx1250) — matrix operand order repaired and checked by reading, not compiled, not run on hardware
//
#include <hip/hip_runtime.h>

// ---------------------------------------------------------------------------
// Types for CDNA5 WMMA (wave32, 16x16x32 bf16 -> f32)
// ---------------------------------------------------------------------------
typedef __attribute__((ext_vector_type(16))) __bf16 v16bf;
typedef __attribute__((ext_vector_type(8)))  float  v8f;

union Frag {              // 32 bytes = one 16-element bf16 WMMA operand
    uint4          q[2];
    unsigned short u[16];
    v16bf          bf;
};

// hardware bf16 convert (clang emits v_cvt_pk_bf16_f32 for pairs)
__device__ __forceinline__ unsigned short f2bf(float f) {
    return __builtin_bit_cast(unsigned short, (__bf16)f);
}

#define ZW   64
#define HID  1024
#define NOUT 12288            // 3*64*64
#define NB   42               // boxes per batch item
#define NL   21               // unique glimpses
#define BSZ  16               // batch
#define MROWS (BSZ*NB)        // 672 = 42 tiles of 16

// ---------------------------------------------------------------------------
// K0: f32 (K x N) -> bf16 transposed (N x K), LDS-tiled 32x32
// ---------------------------------------------------------------------------
__global__ __launch_bounds__(256) void transpose_cvt(
    const float* __restrict__ src, unsigned short* __restrict__ dst,
    int K, int N)
{
    __shared__ unsigned short t[32][33];
    const int nb = blockIdx.x * 32, kb = blockIdx.y * 32;
    #pragma unroll
    for (int j = 0; j < 32; j += 8)
        t[threadIdx.y + j][threadIdx.x] =
            f2bf(src[(size_t)(kb + threadIdx.y + j) * N + nb + threadIdx.x]);
    __syncthreads();
    #pragma unroll
    for (int j = 0; j < 32; j += 8)
        dst[(size_t)(nb + threadIdx.y + j) * K + kb + threadIdx.x] =
            t[threadIdx.x][threadIdx.y + j];
}

// ---------------------------------------------------------------------------
// K1: H = relu(gather(z_what)[672x64] @ w1 + b1), stored bf16 row-major.
// One wave per 16x16 tile; K=64 -> 2 WMMA steps. 42x64 tiles = 2688 waves.
// ---------------------------------------------------------------------------
__global__ __launch_bounds__(256) void gemm1_relu(
    const float* __restrict__ z_what,
    const unsigned short* __restrict__ w1t,    // [HID][ZW] bf16
    const float* __restrict__ b1,
    const int* __restrict__ indices,
    unsigned short* __restrict__ H)            // [672][HID] bf16
{
    const int wave  = (blockIdx.x * blockDim.x + threadIdx.x) >> 5;
    const int lane  = threadIdx.x & 31;
    const int lo    = lane & 15, hi = lane >> 4;
    const int mtile = wave % (MROWS / 16);     // 0..41
    const int ntile = wave / (MROWS / 16);     // 0..63

    // A operand: row M = lo of the gathered z_what matrix
    const int row = mtile * 16 + lo;           // 0..671
    const int bb  = row / NB, nn = row % NB;
    const float* arow = z_what + ((size_t)bb * NL + indices[nn]) * ZW;
    const unsigned short* brow = w1t + (size_t)(ntile * 16 + lo) * ZW;

    v8f acc = {};
    #pragma unroll
    for (int ks = 0; ks < 2; ++ks) {
        const int k0 = ks * 32;
        Frag a, b;
        // A layout: e<8 -> K=k0+hi*8+e ; e>=8 -> K=k0+16+hi*8+(e-8)
        const float4* c0 = (const float4*)(arow + k0 + hi * 8);
        const float4* c1 = (const float4*)(arow + k0 + 16 + hi * 8);
        float4 f0 = c0[0], f1 = c0[1], f2 = c1[0], f3 = c1[1];
        a.bf[0] = (__bf16)f0.x; a.bf[1] = (__bf16)f0.y; a.bf[2]  = (__bf16)f0.z; a.bf[3]  = (__bf16)f0.w;
        a.bf[4] = (__bf16)f1.x; a.bf[5] = (__bf16)f1.y; a.bf[6]  = (__bf16)f1.z; a.bf[7]  = (__bf16)f1.w;
        a.bf[8] = (__bf16)f2.x; a.bf[9] = (__bf16)f2.y; a.bf[10] = (__bf16)f2.z; a.bf[11] = (__bf16)f2.w;
        a.bf[12]= (__bf16)f3.x; a.bf[13]= (__bf16)f3.y; a.bf[14] = (__bf16)f3.z; a.bf[15] = (__bf16)f3.w;
        // B layout: e<8 -> K=k0+hi*8+e ; e>=8 -> K=k0+16+hi*8+(e-8)
        const uint4* bq = (const uint4*)(brow + k0 + hi * 8);
        b.q[0] = bq[0]; b.q[1] = bq[2];
        acc = __builtin_amdgcn_wmma_f32_16x16x32_bf16(
                  false, a.bf, false, b.bf, (short)0, acc, false, false);
    }

    const float bias = b1[ntile * 16 + lo];
    unsigned short* out = H + (size_t)(mtile * 16) * HID + ntile * 16 + lo;
    #pragma unroll
    for (int r = 0; r < 8; ++r) {
        float v = acc[r] + bias;
        v = v > 0.f ? v : 0.f;
        out[(size_t)(r + hi * 8) * HID] = f2bf(v);
    }
}

// ---------------------------------------------------------------------------
// K2: dec = sigmoid(H[672x1024] @ w2 + b2), f32 row-major.
// One wave per 16x64 slab: 4 accumulators, A fragment reused 4x per k-step.
// 4 independent WMMAs/step hide the WMMA->VALU hazard window.
// 42 * 192 = 8064 waves.
// ---------------------------------------------------------------------------
__global__ __launch_bounds__(256) void gemm2_sigmoid(
    const unsigned short* __restrict__ H,      // [672][HID] bf16
    const unsigned short* __restrict__ w2t,    // [NOUT][HID] bf16
    const float* __restrict__ b2,
    float* __restrict__ dec)                   // [672][NOUT] f32
{
    const int wave = (blockIdx.x * blockDim.x + threadIdx.x) >> 5;
    const int lane = threadIdx.x & 31;
    const int lo   = lane & 15, hi = lane >> 4;
    const int mtile = wave % (MROWS / 16);     // 0..41
    const int n0    = (wave / (MROWS / 16)) * 64;  // 0..12224 step 64

    const unsigned short* arow  = H   + (size_t)(mtile * 16 + lo) * HID;
    const unsigned short* brow0 = w2t + (size_t)(n0      + lo) * HID;
    const unsigned short* brow1 = w2t + (size_t)(n0 + 16 + lo) * HID;
    const unsigned short* brow2 = w2t + (size_t)(n0 + 32 + lo) * HID;
    const unsigned short* brow3 = w2t + (size_t)(n0 + 48 + lo) * HID;

    v8f acc0 = {}, acc1 = {}, acc2 = {}, acc3 = {};
    #pragma unroll 2
    for (int k0 = 0; k0 < HID; k0 += 32) {
        Frag a, b0, b1v, b2v, b3v;
        const uint4* a0 = (const uint4*)(arow + k0 + hi * 16);
        a.q[0] = a0[0]; a.q[1] = a0[1];
        const uint4* p0 = (const uint4*)(brow0 + k0 + hi * 16);
        b0.q[0] = p0[0]; b0.q[1] = p0[1];
        const uint4* p1 = (const uint4*)(brow1 + k0 + hi * 16);
        b1v.q[0] = p1[0]; b1v.q[1] = p1[1];
        const uint4* p2 = (const uint4*)(brow2 + k0 + hi * 16);
        b2v.q[0] = p2[0]; b2v.q[1] = p2[1];
        const uint4* p3 = (const uint4*)(brow3 + k0 + hi * 16);
        b3v.q[0] = p3[0]; b3v.q[1] = p3[1];
        // stream-ahead hints for the B operand (emit global_prefetch)
        __builtin_prefetch(brow0 + k0 + 32 + hi * 16, 0, 1);
        __builtin_prefetch(brow2 + k0 + 32 + hi * 16, 0, 1);
        acc0 = __builtin_amdgcn_wmma_f32_16x16x32_bf16(
                   false, a.bf, false, b0.bf, (short)0, acc0, false, false);
        acc1 = __builtin_amdgcn_wmma_f32_16x16x32_bf16(
                   false, a.bf, false, b1v.bf, (short)0, acc1, false, false);
        acc2 = __builtin_amdgcn_wmma_f32_16x16x32_bf16(
                   false, a.bf, false, b2v.bf, (short)0, acc2, false, false);
        acc3 = __builtin_amdgcn_wmma_f32_16x16x32_bf16(
                   false, a.bf, false, b3v.bf, (short)0, acc3, false, false);
    }

    const float bias0 = b2[n0      + lo];
    const float bias1 = b2[n0 + 16 + lo];
    const float bias2 = b2[n0 + 32 + lo];
    const float bias3 = b2[n0 + 48 + lo];
    #pragma unroll
    for (int r = 0; r < 8; ++r) {
        const size_t m = (size_t)(mtile * 16 + r + hi * 8) * NOUT;
        float v0 = acc0[r] + bias0;
        float v1 = acc1[r] + bias1;
        float v2 = acc2[r] + bias2;
        float v3 = acc3[r] + bias3;
        dec[m + n0      + lo] = 1.f / (1.f + __expf(-v0));
        dec[m + n0 + 16 + lo] = 1.f / (1.f + __expf(-v1));
        dec[m + n0 + 32 + lo] = 1.f / (1.f + __expf(-v2));
        dec[m + n0 + 48 + lo] = 1.f / (1.f + __expf(-v3));
    }
}

// ---------------------------------------------------------------------------
// K3: depth softmax weights per batch item (tiny).
// ---------------------------------------------------------------------------
__global__ void softmax_depth(
    const float* __restrict__ z_depth,   // [B][NL][1]
    const int*   __restrict__ z_present, // [B][NB][1]
    const int*   __restrict__ indices,   // [NB]
    float* __restrict__ wgt)             // [B][NB]
{
    const int b = blockIdx.x;
    if (threadIdx.x != 0) return;
    float vals[NB];
    float mx = -__builtin_inff();
    for (int n = 0; n < NB; ++n) {
        float v = (z_present[b * NB + n] == 1)
                ? z_depth[b * NL + indices[n]] : -__builtin_inff();
        vals[n] = v;
        mx = fmaxf(mx, v);
    }
    float s = 0.f;
    for (int n = 0; n < NB; ++n) {
        float e = (vals[n] == -__builtin_inff()) ? 0.f : __expf(vals[n] - mx);
        vals[n] = e;
        s += e;
    }
    const float inv = 1.f / s;
    for (int n = 0; n < NB; ++n) wgt[b * NB + n] = vals[n] * inv;
}

// ---------------------------------------------------------------------------
// K4: bilinear spatial transform + weighted composite.
// grid: (64 pixel-blocks, 3 channels, 16 batch). Box params + weights in LDS.
// ---------------------------------------------------------------------------
__device__ __forceinline__ float samp(const float* __restrict__ img, int xi, int yi) {
    if (xi < 0 || xi > 63 || yi < 0 || yi > 63) return 0.f;
    return img[yi * 64 + xi];
}

__global__ __launch_bounds__(256) void composite(
    const float* __restrict__ dec,       // [672][3][64][64]
    const float* __restrict__ z_where,   // [B][NB][4]
    const float* __restrict__ wgt,       // [B][NB]
    float* __restrict__ out)             // [B][3][128][128]
{
    const int b = blockIdx.z, c = blockIdx.y;
    const int p = blockIdx.x * 256 + threadIdx.x;     // 0..16383
    const int y = p >> 7, x = p & 127;

    __shared__ float s_cx[NB], s_cy[NB], s_w[NB], s_h[NB], s_wt[NB];
    if (threadIdx.x < NB) {
        const int n = threadIdx.x;
        const float* zw = z_where + (size_t)(b * NB + n) * 4;
        s_cx[n] = zw[0]; s_cy[n] = zw[1]; s_w[n] = zw[2]; s_h[n] = zw[3];
        s_wt[n] = wgt[b * NB + n];
    }
    __syncthreads();

    const float xt = (x + 0.5f) * (2.f / 128.f) - 1.f;
    const float yt = (y + 0.5f) * (2.f / 128.f) - 1.f;

    float acc = 0.f;
    for (int n = 0; n < NB; ++n) {
        const float wn = s_wt[n];
        if (wn == 0.f) continue;                         // masked-out box
        const float xs = (xt - (2.f * s_cx[n] - 1.f)) / fmaxf(s_w[n], 1e-6f);
        const float ys = (yt - (2.f * s_cy[n] - 1.f)) / fmaxf(s_h[n], 1e-6f);
        const float px = ((xs + 1.f) * 64.f - 1.f) * 0.5f;
        const float py = ((ys + 1.f) * 64.f - 1.f) * 0.5f;
        if (px <= -1.f || px >= 64.f || py <= -1.f || py >= 64.f) continue;
        const float x0 = floorf(px), y0 = floorf(py);
        const float wx = px - x0,    wy = py - y0;
        const int x0i = (int)x0, y0i = (int)y0;
        const float* img = dec + ((size_t)(b * NB + n) * 3 + c) * 4096;
        const float v00 = samp(img, x0i,     y0i);
        const float v10 = samp(img, x0i + 1, y0i);
        const float v01 = samp(img, x0i,     y0i + 1);
        const float v11 = samp(img, x0i + 1, y0i + 1);
        acc += wn * (v00 * (1.f - wx) * (1.f - wy) + v10 * wx * (1.f - wy) +
                     v01 * (1.f - wx) * wy         + v11 * wx * wy);
    }
    out[((size_t)(b * 3 + c) << 14) + p] = acc;
}

// ---------------------------------------------------------------------------
// Launcher
// ---------------------------------------------------------------------------
extern "C" void kernel_launch(void* const* d_in, const int* in_sizes, int n_in,
                              void* d_out, int out_size, void* d_ws, size_t ws_size,
                              hipStream_t stream) {
    const float* z_what    = (const float*)d_in[0];
    const float* z_where   = (const float*)d_in[1];
    const float* z_depth   = (const float*)d_in[2];
    const float* w1        = (const float*)d_in[3];
    const float* b1        = (const float*)d_in[4];
    const float* w2        = (const float*)d_in[5];
    const float* b2        = (const float*)d_in[6];
    const int*   z_present = (const int*)d_in[7];
    const int*   indices   = (const int*)d_in[8];
    float*       out       = (float*)d_out;

    // workspace layout (all offsets 256B-aligned)
    char* ws = (char*)d_ws;
    unsigned short* w1t = (unsigned short*)ws;                          // 1024*64*2
    unsigned short* w2t = (unsigned short*)(ws + 131072);               // 12288*1024*2
    unsigned short* Hb  = (unsigned short*)(ws + 131072 + 25165824);    // 672*1024*2
    float*          dec = (float*)(ws + 131072 + 25165824 + 1376256);   // 672*12288*4
    float*          wgt = (float*)(ws + 131072 + 25165824 + 1376256 + 33030144);

    // K0: weight conversion/transposition (f32 -> bf16, N-major)
    transpose_cvt<<<dim3(HID / 32, ZW / 32), dim3(32, 8), 0, stream>>>(w1, w1t, ZW, HID);
    transpose_cvt<<<dim3(NOUT / 32, HID / 32), dim3(32, 8), 0, stream>>>(w2, w2t, HID, NOUT);

    // K1: gather + GEMM1 + relu  (42*64 tiles, 8 waves/block)
    gemm1_relu<<<dim3((MROWS / 16) * (HID / 16) / 8), dim3(256), 0, stream>>>(
        z_what, w1t, b1, indices, Hb);

    // K2: GEMM2 + sigmoid  (42*192 slab-waves, 8 waves/block)
    gemm2_sigmoid<<<dim3((MROWS / 16) * (NOUT / 64) / 8), dim3(256), 0, stream>>>(
        Hb, w2t, b2, dec);

    // K3: softmax over depths
    softmax_depth<<<dim3(BSZ), dim3(32), 0, stream>>>(z_depth, z_present, indices, wgt);

    // K4: spatial transform + composite
    composite<<<dim3(64, 3, BSZ), dim3(256), 0, stream>>>(dec, z_where, wgt, out);
}
